// SupervisedRefiner_71554155152237
// MI455X (gfx1250) — matrix operand order repaired and checked by reading, not compiled, not run on hardware
//
#include <hip/hip_runtime.h>
#include <hip/hip_bf16.h>
#include <math.h>

// ---------------------------------------------------------------------------
// SupervisedRefiner for MI455X (gfx1250, wave32, WMMA f16 16x16x32 f32-accum)
//
// Shapes: B=2, N=2048, HID=128, NHEAD=4, HEAD_DIM=32, NLAYER=6, LAT=64, FF=512
//
// d_in layout assumption: d_in[0..7] = the 8 tensors in setup_inputs() dict
// order; d_in[8..] = params pytree leaves flattened with jax sorted-key order:
//   8  final_norm.b      9  final_norm.g
//   10 in_norm.b         11 in_norm.g
//   12 lat_head.l0.b  13 lat_head.l0.w  14 lat_head.l1.b  15 lat_head.l1.w
//   16 + 16*L  (L=0..5): ff.l0.b, ff.l0.w, ff.l1.b, ff.l1.w, k.b, k.w,
//                        n1.b, n1.g, n2.b, n2.g, o.b, o.w, q.b, q.w, v.b, v.w
//   112..115 mech_mlp (l0.b,l0.w,l1.b,l1.w)   116..119 noise_mlp
//   120..123 pos_head                          124..127 pos_mlp
//   128 role_emb                               129..132 time_mlp
// ---------------------------------------------------------------------------

using f16 = _Float16;
typedef __attribute__((ext_vector_type(16))) _Float16 v16h;
typedef __attribute__((ext_vector_type(8)))  _Float16 v8h;
typedef __attribute__((ext_vector_type(8)))  float    v8f;
typedef __attribute__((ext_vector_type(4)))  float    v4f;

#define BB 2
#define NN_ 2048
#define HIDD 128
#define NHEAD_ 4
#define HEADD 32
#define FFD 512
#define LATD 64

#define L2E 1.4426950408889634f

// Fast transcendental helpers: single v_exp_f32 / v_tanh_f32 / v_rcp_f32
// instead of the precise libm expansions (softmax / gates tolerate this).
#if __has_builtin(__builtin_amdgcn_exp2f)
__device__ __forceinline__ float fexp2(float x) { return __builtin_amdgcn_exp2f(x); }
#else
__device__ __forceinline__ float fexp2(float x) { return exp2f(x); }
#endif
__device__ __forceinline__ float fexp(float x) { return fexp2(x * L2E); }
#if __has_builtin(__builtin_amdgcn_tanhf)
__device__ __forceinline__ float ftanh(float x) { return __builtin_amdgcn_tanhf(x); }
#else
__device__ __forceinline__ float ftanh(float x) { return tanhf(x); }
#endif
#if __has_builtin(__builtin_amdgcn_rcpf)
__device__ __forceinline__ float frcp(float x) { return __builtin_amdgcn_rcpf(x); }
#else
__device__ __forceinline__ float frcp(float x) { return 1.0f / x; }
#endif

__device__ __forceinline__ float geluf(float x) {
    return 0.5f * x * (1.0f + erff(x * 0.70710678118654752f));
}

__device__ __forceinline__ v8f wmma16(v16h a, v16h b, v8f c) {
    return __builtin_amdgcn_wmma_f32_16x16x32_f16(
        false, a, false, b, (short)0, c, false, false);
}

// A-operand: 16x32 f16 tile (MxK) from row-major memory, leading dim ld.
// lane m (0..15): K chunks {0..7,16..23}; lane m+16: {8..15,24..31}.
__device__ __forceinline__ v16h load_a(const f16* base, int ld) {
    int lane = threadIdx.x & 31;
    const f16* p = base + (lane & 15) * ld + ((lane >> 4) & 1) * 8;
    v8h x0 = *(const v8h*)(p);
    v8h x1 = *(const v8h*)(p + 16);
    v16h r;
#pragma unroll
    for (int i = 0; i < 8; i++) { r[i] = x0[i]; r[8 + i] = x1[i]; }
    return r;
}

// B-operand: 32x16 f16 tile (KxN). Source rows are the *columns* of B
// (i.e. rows of a transposed matrix), row-major with leading dim ld.
// lane n (0..15): K chunks {0..7,16..23}; lane n+16: {8..15,24..31}.
__device__ __forceinline__ v16h load_b(const f16* base, int ld) {
    int lane = threadIdx.x & 31;
    const f16* p = base + (lane & 15) * ld + ((lane >> 4) & 1) * 8;
    v8h x0 = *(const v8h*)(p);
    v8h x1 = *(const v8h*)(p + 16);
    v16h r;
#pragma unroll
    for (int i = 0; i < 8; i++) { r[i] = x0[i]; r[8 + i] = x1[i]; }
    return r;
}

// ---------------------------------------------------------------------------
// Weight staging: Wt[o*K + k] = (f16) W[k*O + o], zero pad rows o >= O.
// ---------------------------------------------------------------------------
__global__ void k_stage_wt(const float* __restrict__ W, f16* __restrict__ Wt,
                           int K, int O, int Opad) {
    int idx = blockIdx.x * blockDim.x + threadIdx.x;
    if (idx >= Opad * K) return;
    int o = idx / K, k = idx - o * K;
    Wt[idx] = (o < O) ? (f16)W[k * O + o] : (f16)0.0f;
}

// ---------------------------------------------------------------------------
// Pairwise prep: cur_adj (symmetrize, zero-diag, clip), dist, and per-row
// degree / sum(adj*dist) / anchor attachment.
// ---------------------------------------------------------------------------
__global__ void k_prep(const float* __restrict__ pos, const int* __restrict__ roles,
                       const float* __restrict__ A,
                       float* __restrict__ dist, float* __restrict__ curadj,
                       float* __restrict__ aux) {
    const int N = NN_;
    int row = blockIdx.x;              // b*N + i
    int b = row >> 11, i = row & (N - 1);
    int t = threadIdx.x;
    float px = pos[(size_t)row * 3 + 0];
    float py = pos[(size_t)row * 3 + 1];
    float pz = pos[(size_t)row * 3 + 2];
    const float* Ab = A + (size_t)b * N * N;
    const int* rb = roles + b * N;
    float deg = 0.f, se = 0.f, an = 0.f;
    for (int j = t; j < N; j += blockDim.x) {
        float aij = Ab[(size_t)i * N + j];
        float aji = Ab[(size_t)j * N + i];
        float a = 0.5f * (aij + aji);
        a = fminf(fmaxf(a, 0.f), 1.f);
        if (j == i) a = 0.f;
        size_t jo = (size_t)(b * N + j) * 3;
        float dx = px - pos[jo], dy = py - pos[jo + 1], dz = pz - pos[jo + 2];
        float d = sqrtf(dx * dx + dy * dy + dz * dz + 1e-12f);
        size_t o = (size_t)row * N + j;
        curadj[o] = a;
        dist[o] = d;
        deg += a; se += a * d;
        int r = rb[j];
        an += a * ((r == 1 || r == 2) ? 1.f : 0.f);
    }
    __shared__ float s1[256], s2[256], s3[256];
    s1[t] = deg; s2[t] = se; s3[t] = an;
    __syncthreads();
    for (int st = 128; st > 0; st >>= 1) {
        if (t < st) { s1[t] += s1[t + st]; s2[t] += s2[t + st]; s3[t] += s3[t + st]; }
        __syncthreads();
    }
    if (t == 0) {
        aux[row * 3 + 0] = s1[0];
        aux[row * 3 + 1] = s2[0];
        aux[row * 3 + 2] = s3[0];
    }
}

// ---------------------------------------------------------------------------
// Centroids: struct / fixed / mobile per batch -> cents[b*9 + {0..8}]
// ---------------------------------------------------------------------------
__global__ void k_cent(const float* __restrict__ pos, const int* __restrict__ roles,
                       float* __restrict__ cents) {
    const int N = NN_;
    int b = blockIdx.x, t = threadIdx.x;
    float sx = 0, sy = 0, sz = 0, fx = 0, fy = 0, fz = 0, fc = 0;
    float mx = 0, my = 0, mz = 0, mc = 0;
    for (int j = t; j < N; j += blockDim.x) {
        size_t o = (size_t)(b * N + j) * 3;
        float x = pos[o], y = pos[o + 1], z = pos[o + 2];
        sx += x; sy += y; sz += z;
        int r = roles[b * N + j];
        if (r == 1) { fx += x; fy += y; fz += z; fc += 1.f; }
        if (r == 2) { mx += x; my += y; mz += z; mc += 1.f; }
    }
    __shared__ float red[256];
    auto R = [&](float v) -> float {
        red[t] = v; __syncthreads();
        for (int st = 128; st > 0; st >>= 1) {
            if (t < st) red[t] += red[t + st];
            __syncthreads();
        }
        float r = red[0]; __syncthreads();
        return r;
    };
    float SX = R(sx), SY = R(sy), SZ = R(sz);
    float FX = R(fx), FY = R(fy), FZ = R(fz), FC = R(fc);
    float MX = R(mx), MY = R(my), MZ = R(mz), MC = R(mc);
    if (t == 0) {
        cents[b * 9 + 0] = SX / N; cents[b * 9 + 1] = SY / N; cents[b * 9 + 2] = SZ / N;
        float fcc = fmaxf(FC, 1.f), mcc = fmaxf(MC, 1.f);
        cents[b * 9 + 3] = FX / fcc; cents[b * 9 + 4] = FY / fcc; cents[b * 9 + 5] = FZ / fcc;
        cents[b * 9 + 6] = MX / mcc; cents[b * 9 + 7] = MY / mcc; cents[b * 9 + 8] = MZ / mcc;
    }
}

// ---------------------------------------------------------------------------
// Per-batch conditioner vectors: mech + time + noise MLPs -> cond[b][128]
// ---------------------------------------------------------------------------
__global__ void k_condvec(const float* __restrict__ tgt, const float* __restrict__ cur,
                          const float* __restrict__ flow, const float* __restrict__ pn,
                          const float* __restrict__ an,
                          const float* mw0, const float* mb0, const float* mw1, const float* mb1,
                          const float* tw0, const float* tb0, const float* tw1, const float* tb1,
                          const float* nw0, const float* nb0, const float* nw1, const float* nb1,
                          float* __restrict__ cond) {
    int b = blockIdx.x, t = threadIdx.x;   // 128 threads
    __shared__ float f[128], h1[128];
    // ---- mech: 63 triu features ----
    if (t < 21) {
        int r = 0, c = t;
        while (c >= 6 - r) { c -= 6 - r; r++; }
        int i = r, j = r + c;
        float tv = tgt[b * 36 + i * 6 + j];
        float cv = cur[b * 36 + i * 6 + j];
        f[t] = tv; f[21 + t] = cv; f[42 + t] = tv - cv;
    }
    __syncthreads();
    float h = mb0[t];
    for (int k = 0; k < 63; k++) h += f[k] * mw0[k * 128 + t];
    h1[t] = geluf(h);
    __syncthreads();
    float acc = mb1[t];
    for (int k = 0; k < 128; k++) acc += h1[k] * mw1[k * 128 + t];
    __syncthreads();
    // ---- time: sinusoidal(128) ----
    float tv = flow[b];
    if (t < 64) {
        float fr = expf(-6.9077552789821f * (float)t / 63.0f);  // 1000^{-t/63}
        float ang = tv * fr;
        f[t] = sinf(ang);
        f[64 + t] = cosf(ang);
    }
    __syncthreads();
    h = tb0[t];
    for (int k = 0; k < 128; k++) h += f[k] * tw0[k * 128 + t];
    h1[t] = geluf(h);
    __syncthreads();
    float a2 = tb1[t];
    for (int k = 0; k < 128; k++) a2 += h1[k] * tw1[k * 128 + t];
    acc += a2;
    __syncthreads();
    // ---- noise: 3 features ----
    if (t == 0) { f[0] = flow[b]; f[1] = pn[b]; f[2] = an[b]; }
    __syncthreads();
    h = nb0[t];
    for (int k = 0; k < 3; k++) h += f[k] * nw0[k * 128 + t];
    h1[t] = geluf(h);
    __syncthreads();
    float a3 = nb1[t];
    for (int k = 0; k < 128; k++) a3 += h1[k] * nw1[k * 128 + t];
    acc += a3;
    cond[b * 128 + t] = acc;
}

// ---------------------------------------------------------------------------
// Node embedding: [pos,ctx](9) -> pos_mlp -> +role_emb -> LN(in_norm) -> +cond
// One block (128 threads) per node.
// ---------------------------------------------------------------------------
__global__ void k_embed(const float* __restrict__ pos, const int* __restrict__ roles,
                        const float* __restrict__ aux, const float* __restrict__ cents,
                        const float* __restrict__ cond,
                        const float* w0, const float* b0, const float* w1, const float* b1,
                        const float* remb, const float* lng, const float* lnb,
                        float* __restrict__ hidden) {
    int row = blockIdx.x;        // b*N + n
    int b = row >> 11;
    int t = threadIdx.x;
    __shared__ float fin[9], h1[128], red[128];
    if (t == 0) {
        float px = pos[(size_t)row * 3], py = pos[(size_t)row * 3 + 1], pz = pos[(size_t)row * 3 + 2];
        float deg = aux[row * 3], se = aux[row * 3 + 1], an = aux[row * 3 + 2];
        float me = se / fmaxf(deg, 1e-6f);
        const float* C = cents + b * 9;
        float ds = sqrtf((px - C[0]) * (px - C[0]) + (py - C[1]) * (py - C[1]) + (pz - C[2]) * (pz - C[2]) + 1e-12f);
        float df = sqrtf((px - C[3]) * (px - C[3]) + (py - C[4]) * (py - C[4]) + (pz - C[5]) * (pz - C[5]) + 1e-12f);
        float dm = sqrtf((px - C[6]) * (px - C[6]) + (py - C[7]) * (py - C[7]) + (pz - C[8]) * (pz - C[8]) + 1e-12f);
        fin[0] = px; fin[1] = py; fin[2] = pz; fin[3] = deg; fin[4] = me;
        fin[5] = ds; fin[6] = df; fin[7] = dm; fin[8] = an;
    }
    __syncthreads();
    float h = b0[t];
    for (int k = 0; k < 9; k++) h += fin[k] * w0[k * 128 + t];
    h1[t] = geluf(h);
    __syncthreads();
    float h2 = b1[t];
    for (int k = 0; k < 128; k++) h2 += h1[k] * w1[k * 128 + t];
    h2 += remb[roles[row] * 128 + t];
    // layernorm over 128
    red[t] = h2; __syncthreads();
    for (int s = 64; s > 0; s >>= 1) { if (t < s) red[t] += red[t + s]; __syncthreads(); }
    float mean = red[0] * (1.0f / 128.0f); __syncthreads();
    float d = h2 - mean;
    red[t] = d * d; __syncthreads();
    for (int s = 64; s > 0; s >>= 1) { if (t < s) red[t] += red[t + s]; __syncthreads(); }
    float var = red[0] * (1.0f / 128.0f);
    float out = d * rsqrtf(var + 1e-5f) * lng[t] + lnb[t] + cond[b * 128 + t];
    hidden[(size_t)row * 128 + t] = out;
}

// ---------------------------------------------------------------------------
// LayerNorm: hidden(f32)[rows,128] -> f16 normalized output.
// ---------------------------------------------------------------------------
__global__ void k_layernorm(const float* __restrict__ X, const float* __restrict__ g,
                            const float* __restrict__ bb, f16* __restrict__ out) {
    int row = blockIdx.x, t = threadIdx.x;   // 128 threads
    __shared__ float red[128];
    float x = X[(size_t)row * 128 + t];
    red[t] = x; __syncthreads();
    for (int s = 64; s > 0; s >>= 1) { if (t < s) red[t] += red[t + s]; __syncthreads(); }
    float mean = red[0] * (1.0f / 128.0f); __syncthreads();
    float d = x - mean;
    red[t] = d * d; __syncthreads();
    for (int s = 64; s > 0; s >>= 1) { if (t < s) red[t] += red[t + s]; __syncthreads(); }
    float var = red[0] * (1.0f / 128.0f);
    out[(size_t)row * 128 + t] = (f16)(d * rsqrtf(var + 1e-5f) * g[t] + bb[t]);
}

// ---------------------------------------------------------------------------
// Generic WMMA GEMM: out = act(X[M,K] * Wt[NoutPad,K]^T + bias).
// One wave = one 16x16 output tile. ACT: 0=none,1=gelu. RESID: out32 += prev.
// ---------------------------------------------------------------------------
template <int ACT, int RESID>
__global__ void k_gemm(const f16* __restrict__ X, const f16* __restrict__ Wt,
                       const float* __restrict__ bias,
                       float* __restrict__ out32, f16* __restrict__ out16,
                       int M, int K, int NoutPad, int NoutReal, int ldo) {
    int wave = (blockIdx.x * blockDim.x + threadIdx.x) >> 5;
    int lane = threadIdx.x & 31;
    int ntN = NoutPad >> 4;
    int total = (M >> 4) * ntN;
    if (wave >= total) return;
    int tm = wave / ntN, tn = wave - tm * ntN;
    const f16* xb = X + (size_t)tm * 16 * K;
    const f16* wb = Wt + (size_t)tn * 16 * K;
    v8f c{};
    for (int kk = 0; kk < K; kk += 32) {
        if (kk + 32 < K) {   // speculative prefetch of next K-chunk (this lane's row)
            __builtin_prefetch(xb + kk + 32 + (lane & 15) * K, 0, 3);
            __builtin_prefetch(wb + kk + 32 + (lane & 15) * K, 0, 3);
        }
        v16h a = load_a(xb + kk, K);
        v16h b = load_b(wb + kk, K);
        c = wmma16(a, b, c);
    }
    int col = tn * 16 + (lane & 15);
    int hi = (lane >> 4) & 1;
    if (col < NoutReal) {
        float bv = bias ? bias[col] : 0.0f;
#pragma unroll
        for (int v = 0; v < 8; v++) {
            int row = tm * 16 + v + hi * 8;
            float val = c[v] + bv;
            if (ACT == 1) val = geluf(val);
            size_t o = (size_t)row * ldo + col;
            if (RESID) val += out32[o];
            if (out32) out32[o] = val;
            if (out16) out16[o] = (f16)val;
        }
    }
}

// ---------------------------------------------------------------------------
// Transpose V per head: vh[b,n,h*32+d] -> vt[((b*4+h)*32+d)*N + n]  (f16)
// ---------------------------------------------------------------------------
__global__ void k_transpose_v(const f16* __restrict__ vh, f16* __restrict__ vt) {
    const int N = NN_;
    int idx = blockIdx.x * blockDim.x + threadIdx.x;
    if (idx >= BB * N * HIDD) return;
    int c = idx & 127;
    int n = (idx >> 7) & (N - 1);
    int b = idx >> 18;
    int h = c >> 5, d = c & 31;
    vt[((size_t)(b * NHEAD_ + h) * HEADD + d) * N + n] = vh[idx];
}

// ---------------------------------------------------------------------------
// Fused flash-attention, one wave per (b, head, 16-query tile).
// Computes S^T = K * Q^T with WMMA (lane = query column), streaming softmax
// (exp2 domain -> single v_exp_f32 per element) with cond bias/mult, then
// P * V with WMMA (V pre-transposed per head).
// mode: 0 = distance affinity, 1 = connectivity, 2 = free.
// ---------------------------------------------------------------------------
__global__ void k_attn(const f16* __restrict__ Qh, const f16* __restrict__ Kh,
                       const f16* __restrict__ Vt, f16* __restrict__ attOut,
                       const float* __restrict__ dist, const float* __restrict__ adj,
                       int mode) {
    const int N = NN_;
    int w = (blockIdx.x * blockDim.x + threadIdx.x) >> 5;
    int lane = threadIdx.x & 31;
    int qt = w & 127;
    int h = (w >> 7) & 3;
    int b = w >> 9;
    if (b >= BB) return;
    int n15 = lane & 15, hi = (lane >> 4) & 1;
    int qg = qt * 16 + n15;                       // this lane's query
    v16h bq = load_b(Qh + ((size_t)(b * N) + qt * 16) * HIDD + h * HEADD, HIDD);
    const float* cnd = (mode == 0) ? dist : adj;
    const size_t cbase = ((size_t)b * N + qg) * N;
    const f16* kbase = Kh + (size_t)(b * N) * HIDD + h * HEADD;
    const f16* vb = Vt + (size_t)(b * NHEAD_ + h) * HEADD * N;
    const float scale = 0.17677669529663689f;     // 1/sqrt(32)
    float m_run = -1e30f, l_run = 0.0f;
    v8f acc0{}, acc1{};
    for (int t = 0; t < N / 32; ++t) {
        int k0 = t * 32;
        if (t + 1 < N / 32) {   // prefetch next tile (per-lane rows / cond line)
            __builtin_prefetch(kbase + (size_t)(k0 + 32 + n15) * HIDD, 0, 3);
            __builtin_prefetch(vb + (size_t)n15 * N + k0 + 32, 0, 3);
            if (mode < 2) __builtin_prefetch(cnd + cbase + k0 + 32, 0, 3);
        }
        v16h a0 = load_a(kbase + (size_t)k0 * HIDD, HIDD);
        v16h a1 = load_a(kbase + (size_t)(k0 + 16) * HIDD, HIDD);
        v8f z{};
        v8f s0 = wmma16(a0, bq, z);   // S^T tile: keys k0..k0+15  x queries
        v8f s1 = wmma16(a1, bq, z);   // S^T tile: keys k0+16..+31 x queries
        float l0[8], l1[8], mu0[8], mu1[8];
        float lm = -1e30f;
        if (mode < 2) {
            // keys this lane owns are contiguous: k0+hi*8..+7 and +16
            const v4f* cp = (const v4f*)(cnd + cbase + k0 + hi * 8);
            v4f c0a = cp[0], c0b = cp[1];   // keys k0+hi*8 .. +7
            v4f c1a = cp[4], c1b = cp[5];   // keys k0+16+hi*8 .. +7
            float cv[16];
#pragma unroll
            for (int i = 0; i < 4; i++) {
                cv[i] = c0a[i]; cv[4 + i] = c0b[i];
                cv[8 + i] = c1a[i]; cv[12 + i] = c1b[i];
            }
#pragma unroll
            for (int v = 0; v < 8; v++) {
                float c0 = cv[v], c1 = cv[8 + v];
                if (mode == 0) {
                    // exp(-(d/0.22)^2) == exp2(-d^2 * 20.6612 * log2e)
                    c0 = fexp2(-c0 * c0 * 29.807776f);
                    c1 = fexp2(-c1 * c1 * 29.807776f);
                }
                float x0 = s0[v] * scale + 2.f * c0 - 1.f;
                float x1 = s1[v] * scale + 2.f * c1 - 1.f;
                l0[v] = x0; l1[v] = x1;
                mu0[v] = 0.25f + 0.75f * c0;
                mu1[v] = 0.25f + 0.75f * c1;
                lm = fmaxf(lm, fmaxf(x0, x1));
            }
        } else {
#pragma unroll
            for (int v = 0; v < 8; v++) {
                float x0 = s0[v] * scale, x1 = s1[v] * scale;
                l0[v] = x0; l1[v] = x1; mu0[v] = 1.f; mu1[v] = 1.f;
                lm = fmaxf(lm, fmaxf(x0, x1));
            }
        }
        float om = __shfl_xor(lm, 16, 32);
        float m_new = fmaxf(m_run, fmaxf(lm, om));
        float fac = fexp2((m_run - m_new) * L2E);
        float ls = 0.f;
        v16h pw;
#pragma unroll
        for (int v = 0; v < 8; v++) {
            float p0 = fexp2((l0[v] - m_new) * L2E) * mu0[v];
            float p1 = fexp2((l1[v] - m_new) * L2E) * mu1[v];
            ls += p0 + p1;
            pw[v] = (f16)p0;        // tile k0..+15  -> A K-chunk lo
            pw[8 + v] = (f16)p1;    // tile k0+16..  -> A K-chunk hi
        }
        ls += __shfl_xor(ls, 16, 32);
        l_run = l_run * fac + ls;
        m_run = m_new;
        // rescale accumulators (per-row factor broadcast by query index)
#pragma unroll
        for (int v = 0; v < 8; v++) {
            float fv = __shfl(fac, v + hi * 8, 32);
            acc0[v] *= fv;
            acc1[v] *= fv;
        }
        v16h b0 = load_b(vb + k0, N);                    // dims 0..15
        v16h b1 = load_b(vb + (size_t)16 * N + k0, N);   // dims 16..31
        acc0 = wmma16(pw, b0, acc0);
        acc1 = wmma16(pw, b1, acc1);
    }
    float rl = frcp(fmaxf(l_run, 1e-20f));
#pragma unroll
    for (int v = 0; v < 8; v++) {
        float rv = __shfl(rl, v + hi * 8, 32);
        int row = qt * 16 + v + hi * 8;
        size_t o = ((size_t)(b * N) + row) * HIDD + h * HEADD;
        attOut[o + n15] = (f16)(acc0[v] * rv);
        attOut[o + 16 + n15] = (f16)(acc1[v] * rv);
    }
}

// ---------------------------------------------------------------------------
// Final: scores = lat*lat^T / 8, delta=tanh, gate, pred & adj_vel.
// One wave per 16x16 tile of the NxN map.
// ---------------------------------------------------------------------------
__global__ void k_scores(const f16* __restrict__ lath, const float* __restrict__ dist,
                         const float* __restrict__ adj,
                         float* __restrict__ adj_vel, float* __restrict__ pred) {
    const int N = NN_;
    int w = (blockIdx.x * blockDim.x + threadIdx.x) >> 5;
    int lane = threadIdx.x & 31;
    int tj = w & 127;
    int ti = (w >> 7) & 127;
    int b = w >> 14;
    if (b >= BB) return;
    const f16* ab = lath + ((size_t)(b * N) + ti * 16) * LATD;
    const f16* bbp = lath + ((size_t)(b * N) + tj * 16) * LATD;
    v8f c{};
    for (int kk = 0; kk < LATD; kk += 32) {
        v16h a = load_a(ab + kk, LATD);
        v16h bo = load_b(bbp + kk, LATD);
        c = wmma16(a, bo, c);
    }
    int n15 = lane & 15, hi = (lane >> 4) & 1;
    int j = tj * 16 + n15;
#pragma unroll
    for (int v = 0; v < 8; v++) {
        int i = ti * 16 + v + hi * 8;
        size_t o = ((size_t)b * N + i) * N + j;
        float d = dist[o], ca = adj[o];
        float delta = ftanh(c[v] * 0.125f);
        // sigmoid((0.24-d)/0.08): exp2 domain, single trans + rcp
        float g = frcp(1.0f + fexp2(-(0.24f - d) * 12.5f * L2E));
        float p = ca + delta * g - ca * (1.0f - g);
        p = fminf(fmaxf(p, 0.f), 1.f);
        if (i == j) p = 0.f;   // zero diagonal (symmetric by constr.)
        pred[o] = p;
        adj_vel[o] = p - ca;
    }
}

// ---------------------------------------------------------------------------
// Host launch
// ---------------------------------------------------------------------------
extern "C" void kernel_launch(void* const* d_in, const int* in_sizes, int n_in,
                              void* d_out, int out_size, void* d_ws, size_t ws_size,
                              hipStream_t stream) {
    const int B = BB, N = NN_;
    const float* positions = (const float*)d_in[0];
    const int* roles = (const int*)d_in[1];
    const float* adjacency = (const float*)d_in[2];
    const float* tgt = (const float*)d_in[3];
    const float* cur = (const float*)d_in[4];
    const float* flow = (const float*)d_in[5];
    const float* posn = (const float*)d_in[6];
    const float* adjn = (const float*)d_in[7];
#define PW(i) ((const float*)d_in[(i)])

    // ---- workspace carve ----
    size_t off = 0;
    auto carve = [&](size_t bytes) -> char* {
        char* p = (char*)d_ws + off;
        off = (off + bytes + 255) & ~(size_t)255;
        return p;
    };
    const size_t NNt = (size_t)N * N, BN = (size_t)B * N;
    float* dist = (float*)carve(B * NNt * 4);
    float* curadj = (float*)carve(B * NNt * 4);
    float* hidden = (float*)carve(BN * HIDD * 4);
    f16* xh = (f16*)carve(BN * HIDD * 2);
    f16* qh = (f16*)carve(BN * HIDD * 2);
    f16* kh = (f16*)carve(BN * HIDD * 2);
    f16* vh = (f16*)carve(BN * HIDD * 2);
    f16* vt = (f16*)carve((size_t)B * NHEAD_ * HEADD * N * 2);
    f16* att = (f16*)carve(BN * HIDD * 2);
    f16* ffh = (f16*)carve(BN * FFD * 2);
    f16* ph = (f16*)carve(BN * HIDD * 2);
    f16* lath = (f16*)carve(BN * LATD * 2);
    float* aux = (float*)carve(BN * 3 * 4);
    float* cents = (float*)carve(B * 9 * 4);
    float* cond = (float*)carve(B * HIDD * 4);
    // staged transposed f16 weights
    f16* wtq[6]; f16* wtk[6]; f16* wtv[6]; f16* wto[6]; f16* wtf0[6]; f16* wtf1[6];
    for (int L = 0; L < 6; L++) {
        wtq[L] = (f16*)carve(128 * 128 * 2);
        wtk[L] = (f16*)carve(128 * 128 * 2);
        wtv[L] = (f16*)carve(128 * 128 * 2);
        wto[L] = (f16*)carve(128 * 128 * 2);
        wtf0[L] = (f16*)carve(512 * 128 * 2);
        wtf1[L] = (f16*)carve(128 * 512 * 2);
    }
    f16* wtpos0 = (f16*)carve(128 * 128 * 2);
    f16* wtpos1 = (f16*)carve(16 * 128 * 2);   // Nout=3 padded to 16
    f16* wtlat0 = (f16*)carve(128 * 128 * 2);
    f16* wtlat1 = (f16*)carve(64 * 128 * 2);

    auto stage = [&](const float* W, f16* Wt, int K, int O, int Opad) {
        int total = Opad * K;
        k_stage_wt<<<(total + 255) / 256, 256, 0, stream>>>(W, Wt, K, O, Opad);
    };
    for (int L = 0; L < 6; L++) {
        int base = 16 + L * 16;
        stage(PW(base + 13), wtq[L], 128, 128, 128);   // q.w
        stage(PW(base + 5), wtk[L], 128, 128, 128);    // k.w
        stage(PW(base + 15), wtv[L], 128, 128, 128);   // v.w
        stage(PW(base + 11), wto[L], 128, 128, 128);   // o.w
        stage(PW(base + 1), wtf0[L], 128, 512, 512);   // ff.l0.w
        stage(PW(base + 3), wtf1[L], 512, 128, 128);   // ff.l1.w
    }
    stage(PW(121), wtpos0, 128, 128, 128);
    stage(PW(123), wtpos1, 128, 3, 16);
    stage(PW(13), wtlat0, 128, 128, 128);
    stage(PW(15), wtlat1, 128, 64, 64);

    auto gemm = [&](const f16* X, const f16* Wt, const float* bias, float* o32,
                    f16* o16, int M, int K, int NoutPad, int NoutReal, int ldo,
                    int act, int resid) {
        int tiles = (M / 16) * (NoutPad / 16);
        int blocks = (tiles + 7) / 8;
        if (act == 0 && resid == 0)
            k_gemm<0, 0><<<blocks, 256, 0, stream>>>(X, Wt, bias, o32, o16, M, K, NoutPad, NoutReal, ldo);
        else if (act == 1 && resid == 0)
            k_gemm<1, 0><<<blocks, 256, 0, stream>>>(X, Wt, bias, o32, o16, M, K, NoutPad, NoutReal, ldo);
        else
            k_gemm<0, 1><<<blocks, 256, 0, stream>>>(X, Wt, bias, o32, o16, M, K, NoutPad, NoutReal, ldo);
    };

    // ---- preprocessing ----
    k_cent<<<B, 256, 0, stream>>>(positions, roles, cents);
    k_prep<<<B * N, 256, 0, stream>>>(positions, roles, adjacency, dist, curadj, aux);
    k_condvec<<<B, 128, 0, stream>>>(tgt, cur, flow, posn, adjn,
                                     PW(113), PW(112), PW(115), PW(114),
                                     PW(130), PW(129), PW(132), PW(131),
                                     PW(117), PW(116), PW(119), PW(118), cond);
    k_embed<<<B * N, 128, 0, stream>>>(positions, roles, aux, cents, cond,
                                       PW(125), PW(124), PW(127), PW(126),
                                       PW(128), PW(11), PW(10), hidden);

    // ---- transformer layers ----
    int M = B * N;
    for (int L = 0; L < 6; L++) {
        int base = 16 + L * 16;
        int mode = L % 3;   // 0=distance, 1=connectivity, 2=free
        k_layernorm<<<M, 128, 0, stream>>>(hidden, PW(base + 7), PW(base + 6), xh);
        gemm(xh, wtq[L], PW(base + 12), nullptr, qh, M, 128, 128, 128, 128, 0, 0);
        gemm(xh, wtk[L], PW(base + 4), nullptr, kh, M, 128, 128, 128, 128, 0, 0);
        gemm(xh, wtv[L], PW(base + 14), nullptr, vh, M, 128, 128, 128, 128, 0, 0);
        k_transpose_v<<<(B * N * HIDD + 255) / 256, 256, 0, stream>>>(vh, vt);
        k_attn<<<(B * NHEAD_ * (N / 16)) / 8, 256, 0, stream>>>(qh, kh, vt, att, dist, curadj, mode);
        gemm(att, wto[L], PW(base + 10), hidden, nullptr, M, 128, 128, 128, 128, 0, 1);
        k_layernorm<<<M, 128, 0, stream>>>(hidden, PW(base + 9), PW(base + 8), xh);
        gemm(xh, wtf0[L], PW(base + 0), nullptr, ffh, M, 128, 512, 512, 512, 1, 0);
        gemm(ffh, wtf1[L], PW(base + 2), hidden, nullptr, M, 512, 128, 128, 128, 0, 1);
    }

    // ---- heads ----
    float* out = (float*)d_out;
    const size_t OUT_POSVEL = 0;
    const size_t OUT_ADJVEL = (size_t)B * N * 3;
    const size_t OUT_PRED = OUT_ADJVEL + (size_t)B * N * N;
    const size_t OUT_LAT = OUT_PRED + (size_t)B * N * N;

    k_layernorm<<<M, 128, 0, stream>>>(hidden, PW(9), PW(8), xh);   // final_norm
    gemm(xh, wtpos0, PW(120), nullptr, ph, M, 128, 128, 128, 128, 1, 0);
    gemm(ph, wtpos1, PW(122), out + OUT_POSVEL, nullptr, M, 128, 16, 3, 3, 0, 0);
    gemm(xh, wtlat0, PW(12), nullptr, ph, M, 128, 128, 128, 128, 1, 0);
    gemm(ph, wtlat1, PW(14), out + OUT_LAT, lath, M, 128, 64, 64, 64, 0, 0);

    int sc_tiles = B * (N / 16) * (N / 16);
    k_scores<<<(sc_tiles + 7) / 8, 256, 0, stream>>>(lath, dist, curadj,
                                                     out + OUT_ADJVEL, out + OUT_PRED);
    (void)in_sizes; (void)n_in; (void)out_size; (void)ws_size;
#undef PW
}
